// TransformerBlock_58849641889801
// MI455X (gfx1250) — matrix operand order repaired and checked by reading, not compiled, not run on hardware
//
#include <hip/hip_runtime.h>
#include <hip/hip_bf16.h>
#include <math.h>

// ---- problem constants (match reference) ----
#define DD   1024
#define HH   16
#define HDD  64
#define FFD  4096
#define BB   2
#define SS   2048
#define MM   (BB * SS)   // 4096 rows
#define LNEPS 1e-5f

typedef __attribute__((ext_vector_type(16))) __bf16 bf16x16;
typedef __attribute__((ext_vector_type(8)))  __bf16 bf16x8;
typedef __attribute__((ext_vector_type(8)))  float  f32x8;
typedef __attribute__((ext_vector_type(4)))  unsigned int u32x4;
typedef __attribute__((ext_vector_type(4)))  int i32x4;
typedef __attribute__((ext_vector_type(8)))  int i32x8;

#if __has_builtin(__builtin_amdgcn_tensor_load_to_lds)
#define HAVE_TDM 1
#else
#define HAVE_TDM 0
#endif

// ------------------------------------------------------------------
// CDNA5 async global->LDS copy (ASYNCcnt path). Inline asm is portable
// across toolchains. LDS operand = low 32 bits of the generic pointer.
// ------------------------------------------------------------------
static __device__ __forceinline__ void async_load_b128(void* lds_ptr, const void* gptr) {
  const unsigned lds_addr = (unsigned)(unsigned long long)lds_ptr;
  asm volatile("global_load_async_to_lds_b128 %0, %1, off"
               :: "v"(lds_addr), "v"(gptr)
               : "memory");
}
static __device__ __forceinline__ void wait_async0() {
  asm volatile("s_wait_asynccnt 0x0" ::: "memory");
}
static __device__ __forceinline__ void wait_ds0() {
  asm volatile("s_wait_dscnt 0x0" ::: "memory");
}

// ------------------------------------------------------------------
// Tensor Data Mover: 2D tile (tile_d1 rows x tile_d0 elems, bf16) from
// global (row stride = stride_d0 elements) into contiguous LDS.
// D# per CDNA5 ISA. Tensor dims == tile dims
// (tiles are always fully in-bounds here).
// ------------------------------------------------------------------
#if HAVE_TDM
static __device__ __forceinline__ void tdm_load_2d(void* lds_ptr, const void* gptr,
                                                   unsigned tile_d0, unsigned tile_d1,
                                                   unsigned long long stride_d0) {
  const unsigned lds_addr = (unsigned)(unsigned long long)lds_ptr;
  const unsigned long long ga = (unsigned long long)gptr;
  u32x4 g0;
  g0[0] = 1u;                                            // count=1, user D#
  g0[1] = lds_addr;                                      // lds_addr
  g0[2] = (unsigned)(ga & 0xFFFFFFFFu);                  // global_addr[31:0]
  g0[3] = (unsigned)((ga >> 32) & 0x1FFFFFFu) | (2u << 30); // addr[56:32] | type=2
  i32x8 g1;
  const unsigned td0 = tile_d0, td1 = tile_d1;           // tensor dims = tile dims
  g1[0] = (int)(1u << 16);                               // wg_mask=0, data_size=1 (2B)
  g1[1] = (int)((td0 & 0xFFFFu) << 16);                  // tensor_dim0[15:0] @ bit48
  g1[2] = (int)((td0 >> 16) | ((td1 & 0xFFFFu) << 16));  // tensor_dim0 hi | tensor_dim1 lo
  g1[3] = (int)((td1 >> 16) | (tile_d0 << 16));          // tensor_dim1 hi | tile_dim0
  g1[4] = (int)(tile_d1 & 0xFFFFu);                      // tile_dim1 (tile_dim2 = 0)
  g1[5] = (int)(unsigned)(stride_d0 & 0xFFFFFFFFull);    // tensor_dim0_stride[31:0]
  g1[6] = (int)(unsigned)((stride_d0 >> 32) & 0xFFFFull);// stride hi | dim1_stride lo = 0
  g1[7] = 0;
  i32x4 z4 = {0, 0, 0, 0};
#if defined(__clang_major__) && __clang_major__ >= 23
  i32x8 z8 = {0, 0, 0, 0, 0, 0, 0, 0};
  __builtin_amdgcn_tensor_load_to_lds(g0, g1, z4, z4, z8, 0);
#else
  __builtin_amdgcn_tensor_load_to_lds(g0, g1, z4, z4, 0);
#endif
}
#endif

// ------------------------------------------------------------------
// WMMA helpers (v_wmma_f32_16x16x32_bf16, wave32 layouts per ISA)
// ------------------------------------------------------------------
static __device__ __forceinline__ f32x8 wmma_bf16(bf16x16 a, bf16x16 b, f32x8 c) {
  return __builtin_amdgcn_wmma_f32_16x16x32_bf16(false, a, false, b, (short)0, c, false, false);
}

// A-matrix 16x32 bf16 from global: lane = M%16; half-wave selects K+16 group.
static __device__ __forceinline__ bf16x16 load_a_frag(const __bf16* base, int lda, int row0, int k0) {
  const int lane = threadIdx.x & 31;
  const __bf16* p = base + (size_t)(row0 + (lane & 15)) * lda + k0 + ((lane >> 4) << 4);
  bf16x8 lo = *(const bf16x8*)p;
  bf16x8 hi = *(const bf16x8*)(p + 8);
  return __builtin_shufflevector(lo, hi, 0,1,2,3,4,5,6,7,8,9,10,11,12,13,14,15);
}

// B-matrix 32x16 bf16 from a [N,K] (pre-transposed) global buffer, clamped rows.
static __device__ __forceinline__ bf16x16 load_b_frag_clamp(const __bf16* baseT, int ldk,
                                                            int n0, int k0, int nmax) {
  const int lane = threadIdx.x & 31;
  int n = n0 + (lane & 15); if (n > nmax) n = nmax;  // masked later; keep loads in-bounds
  const __bf16* p = baseT + (size_t)n * ldk + k0 + ((lane >> 4) << 4);
  return *(const bf16x16*)p;              // 32B
}

// LDS fragment readers for staged GEMM tiles (k-step = 32 elements/row)
static __device__ __forceinline__ bf16x16 lds_a_frag(const __bf16* sA, int row0) {
  const int lane = threadIdx.x & 31;
  const __bf16* p = sA + (row0 + (lane & 15)) * 32 + ((lane >> 4) << 4);
  bf16x8 lo = *(const bf16x8*)p;
  bf16x8 hi = *(const bf16x8*)(p + 8);
  return __builtin_shufflevector(lo, hi, 0,1,2,3,4,5,6,7,8,9,10,11,12,13,14,15);
}
static __device__ __forceinline__ bf16x16 lds_b_frag(const __bf16* sB, int n0) {
  const int lane = threadIdx.x & 31;
  return *(const bf16x16*)(sB + (n0 + (lane & 15)) * 32 + ((lane >> 4) << 4));
}

// ------------------------------------------------------------------
// fp32 -> bf16 cast with transpose: W[K,N] f32 -> Wt[N,K] bf16
// ------------------------------------------------------------------
__global__ void k_convert_transpose(const float* __restrict__ W, __bf16* __restrict__ Wt,
                                    int K, int N) {
  int idx = blockIdx.x * blockDim.x + threadIdx.x;
  if (idx >= K * N) return;
  int k = idx / N, n = idx - k * N;
  Wt[(size_t)n * K + k] = (__bf16)W[idx];
}

// ------------------------------------------------------------------
// LayerNorm (fp32 in, bf16 out), one block (8 waves) per row
// ------------------------------------------------------------------
__global__ void k_layernorm_bf16(const float* __restrict__ x, const float* __restrict__ g,
                                 const float* __restrict__ b, __bf16* __restrict__ out, int D) {
  __shared__ float s1[256];
  __shared__ float s2[256];
  const int row = blockIdx.x, tid = threadIdx.x;
  const float* xr = x + (size_t)row * D;
  float sum = 0.f, sq = 0.f;
  for (int i = tid; i < D; i += 256) { float v = xr[i]; sum += v; sq += v * v; }
  s1[tid] = sum; s2[tid] = sq; __syncthreads();
  for (int off = 128; off > 0; off >>= 1) {
    if (tid < off) { s1[tid] += s1[tid + off]; s2[tid] += s2[tid + off]; }
    __syncthreads();
  }
  const float mean = s1[0] / (float)D;
  const float var  = s2[0] / (float)D - mean * mean;
  const float rinv = rsqrtf(var + LNEPS);
  __bf16* o = out + (size_t)row * D;
  for (int i = tid; i < D; i += 256)
    o[i] = (__bf16)(g[i] * ((xr[i] - mean) * rinv) + b[i]);
}

// ------------------------------------------------------------------
// bf16 GEMM, TDM-pipelined:  out = act(A @ Wt^T + bias) (+ residual)
// block = 256 threads (8 waves); tile 64(M) x 128(N); k-step 32.
// Wave 0 issues TENSOR_LOAD_TO_LDS for the NEXT k-tile into the spare
// LDS buffer, then s_wait_tensorcnt(2) so the current tile is landed
// while the next streams (TENSORcnt completes in-order). WMMA operands
// come from ds_load_b128. Fallback: async global->LDS staging.
// ------------------------------------------------------------------
template <int ACT, int OUTBF>
__global__ void __launch_bounds__(256)
k_gemm_bf16(const __bf16* __restrict__ A, const __bf16* __restrict__ Wt,
            const float* __restrict__ bias, const float* residual,
            void* outp, int M, int N, int K) {
  __shared__ __bf16 sA[2][64 * 32];    // [buf][m][k]
  __shared__ __bf16 sB[2][128 * 32];   // [buf][n][k]  (Wt is [N,K])

  const int t    = threadIdx.x;
  const int wave = t >> 5;
  const int lane = t & 31;
  const int m0b  = blockIdx.x * 64;
  const int n0b  = blockIdx.y * 128;
  const int m0w  = (wave & 3) * 16;       // wave's rows inside sA
  const int n0w  = (wave >> 2) * 64;      // wave's cols inside sB

  const int nTiles = K >> 5;
  f32x8 acc[4] = {};

#if HAVE_TDM
  if (wave == 0) {                        // prologue: DMA tile 0 -> buf 0
    tdm_load_2d(&sA[0][0], A  + (size_t)m0b * K, 32, 64,  (unsigned long long)K);
    tdm_load_2d(&sB[0][0], Wt + (size_t)n0b * K, 32, 128, (unsigned long long)K);
  }
  for (int i = 0; i < nTiles; ++i) {
    const int cur = i & 1;
    if (wave == 0) {
      if (i + 1 < nTiles) {               // stream next tile into spare buffer
        const int k0n = (i + 1) << 5;
        tdm_load_2d(&sA[cur ^ 1][0], A  + (size_t)m0b * K + k0n, 32, 64,  (unsigned long long)K);
        tdm_load_2d(&sB[cur ^ 1][0], Wt + (size_t)n0b * K + k0n, 32, 128, (unsigned long long)K);
        __builtin_amdgcn_s_wait_tensorcnt(2);  // 2 in flight allowed; tile i landed
      } else {
        __builtin_amdgcn_s_wait_tensorcnt(0);
      }
    }
    __syncthreads();                      // tile i visible to all waves
    const bf16x16 a = lds_a_frag(&sA[cur][0], m0w);
#pragma unroll
    for (int nt = 0; nt < 4; ++nt)
      acc[nt] = wmma_bf16(a, lds_b_frag(&sB[cur][0], n0w + nt * 16), acc[nt]);
    __syncthreads();                      // done reading buf before it is re-DMAed
  }
#else
  const int srow = t >> 2;                // 0..63
  const int sseg = (t & 3) << 3;          // 0,8,16,24 elements
  for (int i = 0; i < nTiles; ++i) {
    const int k0 = i << 5;
    if (k0) __syncthreads();
    async_load_b128(&sA[0][srow * 32 + sseg], A + (size_t)(m0b + srow) * K + k0 + sseg);
    async_load_b128(&sB[0][srow * 32 + sseg], Wt + (size_t)(n0b + srow) * K + k0 + sseg);
    async_load_b128(&sB[0][(srow + 64) * 32 + sseg], Wt + (size_t)(n0b + 64 + srow) * K + k0 + sseg);
    wait_async0();
    __syncthreads();
    const bf16x16 a = lds_a_frag(&sA[0][0], m0w);
#pragma unroll
    for (int nt = 0; nt < 4; ++nt)
      acc[nt] = wmma_bf16(a, lds_b_frag(&sB[0][0], n0w + nt * 16), acc[nt]);
  }
#endif

  // C/D layout: lane%16 = N, VGPR r -> M = r + 8*(lane/16)
  const int col = lane & 15;
  const int mhi = (lane >> 4) << 3;
#pragma unroll
  for (int nt = 0; nt < 4; ++nt) {
    const int n = n0b + n0w + nt * 16 + col;
    const float bv = bias ? bias[n] : 0.f;
#pragma unroll
    for (int r = 0; r < 8; ++r) {
      const int m = m0b + m0w + mhi + r;
      float v = acc[nt][r] + bv;
      if (ACT == 1) v = 0.5f * v * (1.f + erff(v * 0.70710678118654752f));
      if (residual) v += residual[(size_t)m * N + n];
      if (OUTBF) ((__bf16*)outp)[(size_t)m * N + n] = (__bf16)v;
      else       ((float*)outp)[(size_t)m * N + n]  = v;
    }
  }
}

// ------------------------------------------------------------------
// Flash attention (causal), bf16 in/out, fp32 online softmax state.
// grid = (B*H, S/64), block = 128 (4 waves); each wave owns 16 query rows.
// V tiles arrive via async global->LDS (wave-private, no barrier);
// P transposes C-layout -> A-layout through LDS.
// ------------------------------------------------------------------
#define VSTRIDE 72   // 144B rows: 16B-aligned staging, conflict-spread column reads
#define PSTRIDE 40   // 80B rows: 16B-aligned fragment reloads

__global__ void __launch_bounds__(128)
k_attention(const __bf16* __restrict__ Q, const __bf16* __restrict__ Km,
            const __bf16* __restrict__ V, __bf16* __restrict__ O) {
  __shared__ __bf16 ldsV[4][32 * VSTRIDE];
  __shared__ __bf16 ldsP[4][16 * PSTRIDE];

  const int w    = threadIdx.x >> 5;
  const int lane = threadIdx.x & 31;
  const int b    = blockIdx.x / HH;
  const int h    = blockIdx.x % HH;
  const int q0   = (blockIdx.y * 4 + w) * 16;

  const __bf16* Qh = Q  + (size_t)b * SS * DD + h * HDD;
  const __bf16* Kh = Km + (size_t)b * SS * DD + h * HDD;
  const __bf16* Vh = V  + (size_t)b * SS * DD + h * HDD;

  const bf16x16 aq0 = load_a_frag(Qh, DD, q0, 0);
  const bf16x16 aq1 = load_a_frag(Qh, DD, q0, 32);

  f32x8 acc[4] = {};
  float mrun[8], lrun[8];
#pragma unroll
  for (int r = 0; r < 8; ++r) { mrun[r] = -__builtin_inff(); lrun[r] = 0.f; }

  const int col  = lane & 15;
  const int mhi  = (lane >> 4) << 3;
  const int kend = q0 + 15;               // last key needed (causal)

  for (int kk = 0; kk <= kend; kk += 32) {
    // ---- async-DMA the V tile [32 keys x 64 dims] into this wave's LDS ----
    {
      const int key = kk + lane;
      if (key < SS) {
        const __bf16* src = Vh + (size_t)key * DD;
        __bf16* dst = &ldsV[w][lane * VSTRIDE];
#pragma unroll
        for (int j = 0; j < 8; ++j)
          async_load_b128(dst + j * 8, src + j * 8);
      }
    }

    // ---- scores: two 16x16 tiles, S = Q K^T (hd=64 -> two k=32 wmma each) ----
    f32x8 c0 = {}, c1 = {};
    c0 = wmma_bf16(aq0, load_b_frag_clamp(Kh, DD, kk,      0,  SS - 1), c0);
    c0 = wmma_bf16(aq1, load_b_frag_clamp(Kh, DD, kk,      32, SS - 1), c0);
    c1 = wmma_bf16(aq0, load_b_frag_clamp(Kh, DD, kk + 16, 0,  SS - 1), c1);
    c1 = wmma_bf16(aq1, load_b_frag_clamp(Kh, DD, kk + 16, 32, SS - 1), c1);

    // ---- scale, causal mask, online softmax; park P (bf16) in LDS ----
#pragma unroll
    for (int r = 0; r < 8; ++r) {
      const int q = q0 + mhi + r;
      float s0 = c0[r] * 0.125f; if (kk      + col > q) s0 = -__builtin_inff();
      float s1 = c1[r] * 0.125f; if (kk + 16 + col > q) s1 = -__builtin_inff();
      float mx = fmaxf(s0, s1);
#pragma unroll
      for (int d = 1; d < 16; d <<= 1) mx = fmaxf(mx, __shfl_xor(mx, d, 32));
      const float mnew = fmaxf(mrun[r], mx);
      const float p0 = __expf(s0 - mnew);
      const float p1 = __expf(s1 - mnew);
      float rs = p0 + p1;
#pragma unroll
      for (int d = 1; d < 16; d <<= 1) rs += __shfl_xor(rs, d, 32);
      const float corr = __expf(mrun[r] - mnew);
      lrun[r] = lrun[r] * corr + rs;
      mrun[r] = mnew;
#pragma unroll
      for (int nt = 0; nt < 4; ++nt) acc[nt][r] *= corr;
      ldsP[w][(mhi + r) * PSTRIDE + col]      = (__bf16)p0;
      ldsP[w][(mhi + r) * PSTRIDE + 16 + col] = (__bf16)p1;
    }

    // wave-private LDS: drain our ds stores and the async V DMA before reuse
    wait_ds0();
    wait_async0();

    // ---- reload P as an A-fragment (16x32), V columns as B-fragments ----
    const __bf16* pp = &ldsP[w][(lane & 15) * PSTRIDE + ((lane >> 4) << 4)];
    bf16x8 plo = *(const bf16x8*)pp;
    bf16x8 phi = *(const bf16x8*)(pp + 8);
    bf16x16 pa = __builtin_shufflevector(plo, phi, 0,1,2,3,4,5,6,7,8,9,10,11,12,13,14,15);

#pragma unroll
    for (int nt = 0; nt < 4; ++nt) {
      bf16x16 bv;
      const __bf16* vp = &ldsV[w][(((lane >> 4) << 4)) * VSTRIDE + nt * 16 + col];
#pragma unroll
      for (int e = 0; e < 16; ++e) bv[e] = vp[e * VSTRIDE];
      acc[nt] = wmma_bf16(pa, bv, acc[nt]);
    }
  }

  // ---- normalize and store attended (bf16) ----
#pragma unroll
  for (int nt = 0; nt < 4; ++nt) {
#pragma unroll
    for (int r = 0; r < 8; ++r) {
      const int q = q0 + mhi + r;
      const float v = acc[nt][r] / lrun[r];
      O[((size_t)b * SS + q) * DD + h * HDD + nt * 16 + col] = (__bf16)v;
    }
  }
}

// ------------------------------------------------------------------
// host-side orchestration
// ------------------------------------------------------------------
extern "C" void kernel_launch(void* const* d_in, const int* in_sizes, int n_in,
                              void* d_out, int out_size, void* d_ws, size_t ws_size,
                              hipStream_t stream) {
  (void)in_sizes; (void)n_in; (void)out_size; (void)ws_size;
  const float* x    = (const float*)d_in[0];
  const float* ln1g = (const float*)d_in[1];
  const float* ln1b = (const float*)d_in[2];
  const float* ln2g = (const float*)d_in[3];
  const float* ln2b = (const float*)d_in[4];
  const float* Wq = (const float*)d_in[5];  const float* bq = (const float*)d_in[6];
  const float* Wk = (const float*)d_in[7];  const float* bk = (const float*)d_in[8];
  const float* Wv = (const float*)d_in[9];  const float* bv = (const float*)d_in[10];
  const float* Wo = (const float*)d_in[11]; const float* bo = (const float*)d_in[12];
  const float* W1 = (const float*)d_in[13]; const float* b1 = (const float*)d_in[14];
  const float* W2 = (const float*)d_in[15]; const float* b2 = (const float*)d_in[16];
  float* out = (float*)d_out;

  char* ws = (char*)d_ws;
  size_t off = 0;
  auto wsalloc = [&](size_t bytes) -> void* {
    void* p = ws + off;
    off = (off + bytes + 255) & ~(size_t)255;
    return p;
  };
  __bf16* hln  = (__bf16*)wsalloc((size_t)MM * DD * 2);
  __bf16* Wqt  = (__bf16*)wsalloc((size_t)DD * DD * 2);
  __bf16* Wkt  = (__bf16*)wsalloc((size_t)DD * DD * 2);
  __bf16* Wvt  = (__bf16*)wsalloc((size_t)DD * DD * 2);
  __bf16* Wot  = (__bf16*)wsalloc((size_t)DD * DD * 2);
  __bf16* W1t  = (__bf16*)wsalloc((size_t)DD * FFD * 2);
  __bf16* W2t  = (__bf16*)wsalloc((size_t)FFD * DD * 2);
  __bf16* Qb   = (__bf16*)wsalloc((size_t)MM * DD * 2);
  __bf16* Kb   = (__bf16*)wsalloc((size_t)MM * DD * 2);
  __bf16* Vb   = (__bf16*)wsalloc((size_t)MM * DD * 2);
  __bf16* attn = (__bf16*)wsalloc((size_t)MM * DD * 2);
  __bf16* h2   = (__bf16*)wsalloc((size_t)MM * DD * 2);
  __bf16* f1   = (__bf16*)wsalloc((size_t)MM * FFD * 2);

  // weights: fp32 -> bf16, transposed so B-fragments are contiguous K runs
  auto ct = [&](const float* W, __bf16* Wt, int K, int N) {
    int total = K * N;
    k_convert_transpose<<<(total + 255) / 256, 256, 0, stream>>>(W, Wt, K, N);
  };
  ct(Wq, Wqt, DD, DD);  ct(Wk, Wkt, DD, DD);  ct(Wv, Wvt, DD, DD);
  ct(Wo, Wot, DD, DD);  ct(W1, W1t, DD, FFD); ct(W2, W2t, FFD, DD);

  const dim3 gD(MM / 64, DD / 128);     // N = 1024
  const dim3 gF(MM / 64, FFD / 128);    // N = 4096

  // --- attention sub-block ---
  k_layernorm_bf16<<<MM, 256, 0, stream>>>(x, ln1g, ln1b, hln, DD);
  k_gemm_bf16<0, 1><<<gD, 256, 0, stream>>>(hln, Wqt, bq, nullptr, Qb, MM, DD, DD);
  k_gemm_bf16<0, 1><<<gD, 256, 0, stream>>>(hln, Wkt, bk, nullptr, Kb, MM, DD, DD);
  k_gemm_bf16<0, 1><<<gD, 256, 0, stream>>>(hln, Wvt, bv, nullptr, Vb, MM, DD, DD);
  k_attention<<<dim3(BB * HH, SS / 64), 128, 0, stream>>>(Qb, Kb, Vb, attn);
  k_gemm_bf16<0, 0><<<gD, 256, 0, stream>>>(attn, Wot, bo, x, out, MM, DD, DD);

  // --- FFN sub-block ---
  k_layernorm_bf16<<<MM, 256, 0, stream>>>(out, ln2g, ln2b, h2, DD);
  k_gemm_bf16<1, 1><<<gF, 256, 0, stream>>>(h2, W1t, b1, nullptr, f1, MM, FFD, DD);
  k_gemm_bf16<0, 0><<<gD, 256, 0, stream>>>(f1, W2t, b2, out, out, MM, DD, FFD);
}
